// DGCNN_scan_8761733284516
// MI455X (gfx1250) — matrix operand order repaired and checked by reading, not compiled, not run on hardware
//
#include <hip/hip_runtime.h>

// ---------------------------------------------------------------------------
// Types / helpers
// ---------------------------------------------------------------------------
typedef __bf16 bf16_t;
typedef __attribute__((ext_vector_type(16))) __bf16 v16bf;
typedef __attribute__((ext_vector_type(8)))  __bf16 v8bf;
typedef __attribute__((ext_vector_type(8)))  float  v8f;
typedef __attribute__((ext_vector_type(2)))  float  f2v;
typedef __attribute__((ext_vector_type(2)))  unsigned u2v;
typedef __attribute__((ext_vector_type(4)))  unsigned u4v;

#define BN_EPS 1e-5f
#define LRELU_SLOPE 0.2f
#define BN_RSQRT 0.99999500f   // rsqrt(1 + 1e-5)

__device__ __forceinline__ unsigned short bfb(float f) {   // f32 -> bf16 bits (RNE)
  unsigned u = __builtin_bit_cast(unsigned, f);
  return (unsigned short)((u + 0x7FFFu + ((u >> 16) & 1u)) >> 16);
}
__device__ __forceinline__ unsigned pack2(float a, float b) {
  return (unsigned)bfb(a) | ((unsigned)bfb(b) << 16);
}
// order-preserving float <-> uint map (for atomic max reductions)
__device__ __forceinline__ unsigned encf(float f) {
  unsigned u = __builtin_bit_cast(unsigned, f);
  return (u >> 31) ? ~u : (u | 0x80000000u);
}
__device__ __forceinline__ float decf(unsigned e) {
  unsigned u = (e >> 31) ? (e & 0x7FFFFFFFu) : ~e;
  return __builtin_bit_cast(float, u);
}
// build v16bf fragment from two contiguous 8x bf16 LDS runs (2x ds_load_b128)
__device__ __forceinline__ v16bf frag2(const bf16_t* row, int kb, int step) {
  v8bf lo = *(const v8bf*)(row + kb);
  v8bf hi = *(const v8bf*)(row + kb + step);
  return __builtin_shufflevector(lo, hi, 0,1,2,3,4,5,6,7,8,9,10,11,12,13,14,15);
}

// ---------------------------------------------------------------------------
// Fused WMMA GEMM:  Y = epilogue( W (Cout x Cin) * X (Cin x Ncols) )
//   XMODE 0: plain matrix     XMODE 1: on-the-fly edge features (kNN gather)
//   XMODE 2: concat( src1 [opt broadcast over cols], src2 )
//   EPI 0: store (+concat offset)   EPI 1: atomic max over k-groups of cols
//   EPI 2: atomic max over all cols (global max)
// Block: 256 thr = 8 waves, 64x64 tile, double-buffered LDS, async W tiles.
// ---------------------------------------------------------------------------
struct GemmP {
  const unsigned short* W;        // bf16 weights (Cout x Cin)
  const float* gamma; const float* beta;
  int Cout, Cin, Ncols;
  const float* X0; int x0Bs;      // plain / cat2-src1 / edge xsrc
  const float* X1; int x1Bs;      // cat2-src2
  const int* idx; int kk; int Nq; // edge mode
  int C1, C2, bc1;                // cat2
  int Cedge, Npts;                // edge mode
  float* out; unsigned* uout;
  int outCB, chOff, outN;
};

template<bool GUARD>
__device__ __forceinline__ void stage_W(const GemmP& p, int rowBase, int k0,
                                        bf16_t (*Wl)[40], int tid) {
  int r = tid >> 2, kb = (tid & 3) * 8;
  if (!GUARD) {
    // 16B per lane, raw bf16: async DMA global -> LDS (ASYNCcnt tracked)
    const unsigned short* gp = p.W + (size_t)(rowBase + r) * p.Cin + k0 + kb;
    unsigned long long ga = (unsigned long long)gp;
    unsigned la = (unsigned)(size_t)&Wl[r][kb];
    asm volatile("global_load_async_to_lds_b128 %0, %1, off"
                 :: "v"(la), "v"(ga) : "memory");
  } else {
    #pragma unroll
    for (int i = 0; i < 8; ++i) {
      int k = k0 + kb + i;
      unsigned short h = (k < p.Cin) ? p.W[(size_t)(rowBase + r) * p.Cin + k]
                                     : (unsigned short)0;
      Wl[r][kb + i] = __builtin_bit_cast(bf16_t, h);
    }
  }
}

template<int XM, bool GUARD>
__device__ __forceinline__ void stage_X(const GemmP& p, int b, int colBase, int k0,
                                        bf16_t (*Xl)[40], int tid) {
  int kq = (tid >> 5) * 4;          // 0,4,...,28
  int c0 = (tid & 31) * 2;          // 0..62
  float v[2][4];
  if (XM == 0) {
    const float* xb = p.X0 + (size_t)b * p.x0Bs;
    #pragma unroll
    for (int i = 0; i < 4; ++i) {
      int c = k0 + kq + i;
      if (!GUARD || c < p.Cin) {
        f2v t = *(const f2v*)(xb + (size_t)c * p.Ncols + colBase + c0);
        v[0][i] = t.x; v[1][i] = t.y;
      } else { v[0][i] = 0.f; v[1][i] = 0.f; }
    }
  } else if (XM == 1) {
    const float* xs = p.X0 + (size_t)b * p.x0Bs;
    #pragma unroll
    for (int ci = 0; ci < 2; ++ci) {
      int col = colBase + c0 + ci;
      int n = col / p.kk, j = col - n * p.kk;
      int m = p.idx[((size_t)b * p.Nq + n) * p.kk + j];
      #pragma unroll
      for (int i = 0; i < 4; ++i) {
        int c = k0 + kq + i;
        float val = 0.f;
        if (c < p.Cedge)          val = xs[(size_t)c * p.Npts + m] - xs[(size_t)c * p.Npts + n];
        else if (c < 2 * p.Cedge) val = xs[(size_t)(c - p.Cedge) * p.Npts + n];
        v[ci][i] = val;
      }
    }
  } else {
    #pragma unroll
    for (int i = 0; i < 4; ++i) {
      int c = k0 + kq + i;
      if (c < p.C1) {
        if (p.bc1) {
          float t = p.X0[(size_t)b * p.x0Bs + c];
          v[0][i] = t; v[1][i] = t;
        } else {
          f2v t = *(const f2v*)(p.X0 + (size_t)b * p.x0Bs + (size_t)c * p.Ncols + colBase + c0);
          v[0][i] = t.x; v[1][i] = t.y;
        }
      } else if (c < p.C1 + p.C2) {
        f2v t = *(const f2v*)(p.X1 + (size_t)b * p.x1Bs + (size_t)(c - p.C1) * p.Ncols + colBase + c0);
        v[0][i] = t.x; v[1][i] = t.y;
      } else { v[0][i] = 0.f; v[1][i] = 0.f; }
    }
  }
  #pragma unroll
  for (int ci = 0; ci < 2; ++ci) {
    u2v pk = { pack2(v[ci][0], v[ci][1]), pack2(v[ci][2], v[ci][3]) };
    *(u2v*)(&Xl[c0 + ci][kq]) = pk;   // ds_store_b64
  }
}

template<int XM, int EPI, bool LRELU>
__global__ __launch_bounds__(256) void gemm_wmma(GemmP p) {
  __shared__ __align__(16) bf16_t Wl[2][64][40];   // 64 rows x 32 K, ping-pong
  __shared__ __align__(16) bf16_t Xl[2][64][40];   // 64 cols x 32 K, ping-pong

  const int b = blockIdx.z;
  const int rowBase = blockIdx.y * 64;
  const int colBase = blockIdx.x * 64;
  const int tid  = threadIdx.x;
  const int lane = tid & 31, wave = tid >> 5;
  const int wr  = (wave >> 1) << 4;       // wave row tile (0,16,32,48)
  const int wc0 = (wave & 1) * 32;        // wave col tiles (0/32) and +16
  const int wc1 = wc0 + 16;
  const int half = lane >> 4, l16 = lane & 15;

  v8f acc0, acc1;
  #pragma unroll
  for (int i = 0; i < 8; ++i) { acc0[i] = 0.f; acc1[i] = 0.f; }

  const int nT = (p.Cin + 31) >> 5;
  if (32 <= p.Cin) {
    stage_W<false>(p, rowBase, 0, Wl[0], tid);
    stage_X<XM, false>(p, b, colBase, 0, Xl[0], tid);
  } else {
    stage_W<true>(p, rowBase, 0, Wl[0], tid);
    stage_X<XM, true>(p, b, colBase, 0, Xl[0], tid);
  }
  asm volatile("s_wait_asynccnt 0x0" ::: "memory");
  __syncthreads();

  for (int t = 0; t < nT; ++t) {
    const int cur = t & 1;
    if (t + 1 < nT) {               // prefetch next tile into other buffer
      int k0 = (t + 1) << 5;
      if (k0 + 32 <= p.Cin) {
        stage_W<false>(p, rowBase, k0, Wl[cur ^ 1], tid);
        stage_X<XM, false>(p, b, colBase, k0, Xl[cur ^ 1], tid);
      } else {
        stage_W<true>(p, rowBase, k0, Wl[cur ^ 1], tid);
        stage_X<XM, true>(p, b, colBase, k0, Xl[cur ^ 1], tid);
      }
    }
    v16bf af  = frag2(&Wl[cur][wr  + l16][0], half * 8, 16);
    v16bf bf0 = frag2(&Xl[cur][wc0 + l16][0], half * 8, 16);
    v16bf bf1 = frag2(&Xl[cur][wc1 + l16][0], half * 8, 16);
    acc0 = __builtin_amdgcn_wmma_f32_16x16x32_bf16(false, af, false, bf0,
                                                   (short)0, acc0, false, false);
    acc1 = __builtin_amdgcn_wmma_f32_16x16x32_bf16(false, af, false, bf1,
                                                   (short)0, acc1, false, false);
    asm volatile("s_wait_asynccnt 0x0" ::: "memory");
    __syncthreads();
  }

  const int col0 = colBase + wc0 + l16;
  const int col1 = colBase + wc1 + l16;
  int n0 = 0, n1 = 0;
  if (EPI == 1) { n0 = col0 / p.kk; n1 = col1 / p.kk; }
  #pragma unroll
  for (int r = 0; r < 8; ++r) {
    int o = rowBase + wr + (half ? 8 + r : r);
    float s  = p.gamma ? p.gamma[o] * BN_RSQRT : 1.f;
    float bb = p.beta ? p.beta[o] : 0.f;
    float y0 = acc0[r] * s + bb;
    float y1 = acc1[r] * s + bb;
    if (LRELU) {
      y0 = (y0 >= 0.f) ? y0 : LRELU_SLOPE * y0;
      y1 = (y1 >= 0.f) ? y1 : LRELU_SLOPE * y1;
    }
    if (EPI == 0) {
      float* ob = p.out + ((size_t)b * p.outCB + p.chOff + o) * p.Ncols;
      ob[col0] = y0; ob[col1] = y1;
    } else if (EPI == 1) {
      unsigned* ub = p.uout + ((size_t)b * p.Cout + o) * p.outN;
      atomicMax(&ub[n0], encf(y0));
      atomicMax(&ub[n1], encf(y1));
    } else {
      unsigned* ub = p.uout + (size_t)b * p.Cout + o;
      atomicMax(ub, encf(y0));
      atomicMax(ub, encf(y1));
    }
  }
}

// ---------------------------------------------------------------------------
// weight pre-conversion f32 -> bf16
// ---------------------------------------------------------------------------
__global__ void cvt_bf16_kernel(const float* src, unsigned short* dst, int n) {
  int i = blockIdx.x * blockDim.x + threadIdx.x;
  if (i < n) dst[i] = bfb(src[i]);
}

// ---------------------------------------------------------------------------
// decode atomic-max uint buffer -> float (with channel-concat placement)
// ---------------------------------------------------------------------------
__global__ void decode_kernel(const unsigned* u, float* dst, int C, int N,
                              int dstCB, int chOff) {
  int t = blockIdx.x * blockDim.x + threadIdx.x;
  int b = blockIdx.y;
  if (t >= C * N) return;
  int c = t / N, n = t - c * N;
  dst[((size_t)b * dstCB + chOff + c) * N + n] = decf(u[((size_t)b * C + c) * N + n]);
}

// ---------------------------------------------------------------------------
// kNN (squared distance, top-k smallest, ties -> earliest index)
// ---------------------------------------------------------------------------
__global__ void knn_kernel(const float* Q, int qBs, const float* P, int pBs,
                           int C, int Nq, int Np, int k, int* idx) {
  int t = blockIdx.x * blockDim.x + threadIdx.x;
  int b = blockIdx.y;
  if (t >= Nq) return;
  const float* q  = Q + (size_t)b * qBs;
  const float* pp = P + (size_t)b * pBs;
  float bd[20]; int bi[20];
  for (int i = 0; i < k; ++i) { bd[i] = 3.4e38f; bi[i] = 0; }
  for (int m = 0; m < Np; ++m) {
    __builtin_prefetch(&pp[m + 32], 0, 1);
    float d = 0.f;
    for (int c = 0; c < C; ++c) {
      float df = q[(size_t)c * Nq + t] - pp[(size_t)c * Np + m];
      d += df * df;
    }
    if (d < bd[k - 1]) {
      int pos = k - 1;
      while (pos > 0 && bd[pos - 1] > d) {
        bd[pos] = bd[pos - 1]; bi[pos] = bi[pos - 1]; --pos;
      }
      bd[pos] = d; bi[pos] = m;
    }
  }
  for (int i = 0; i < k; ++i) idx[((size_t)b * Nq + t) * k + i] = bi[i];
}

// ---------------------------------------------------------------------------
// Pool: attention scores, per-branch top-k, gather
// ---------------------------------------------------------------------------
__global__ void scores_kernel(const float* xt1, const float* vecp, float* sc, int N) {
  int t = blockIdx.x * blockDim.x + threadIdx.x;
  int b = blockIdx.y;
  if (t >= N * 8) return;
  int n = t >> 3, e = t & 7;
  const float* xb = xt1 + (size_t)b * 128 * N;
  const float* vb = vecp + (size_t)b * 1024;
  float s = 0.f;
  for (int c = 0; c < 128; ++c) s += xb[(size_t)c * N + n] * vb[c * 8 + e];
  sc[((size_t)b * N + n) * 8 + e] = 1.f / (1.f + __expf(-s));
}

__global__ void pool_topk_kernel(const float* sc, int* pidx, float* pval, int N, int R) {
  int e = threadIdx.x;      // 0..7
  int b = blockIdx.x;
  float bv[64]; int bi[64];
  for (int i = 0; i < R; ++i) { bv[i] = -3.4e38f; bi[i] = 0; }
  for (int n = 0; n < N; ++n) {
    float v = sc[((size_t)b * N + n) * 8 + e];
    if (v > bv[R - 1]) {
      int pos = R - 1;
      while (pos > 0 && bv[pos - 1] < v) {
        bv[pos] = bv[pos - 1]; bi[pos] = bi[pos - 1]; --pos;
      }
      bv[pos] = v; bi[pos] = n;
    }
  }
  for (int i = 0; i < R; ++i) {
    int j = i * 8 + e;
    pidx[b * (R * 8) + j] = bi[i];
    pval[b * (R * 8) + j] = bv[i];
  }
}

__global__ void pool_gather_kernel(const float* xyz, const float* xt1,
                                   const int* pidx, const float* pval,
                                   float* node_out, float* nodestat_out,
                                   float* fbuf, int N, int Mp) {
  int j = blockIdx.x * blockDim.x + threadIdx.x;
  int b = blockIdx.y;
  if (j >= Mp) return;
  int id = pidx[b * Mp + j];
  float v = pval[b * Mp + j];
  for (int d = 0; d < 3; ++d) {
    float xs = xyz[((size_t)b * 3 + d) * N + id];
    nodestat_out[((size_t)b * 3 + d) * Mp + j] = xs;
    node_out[((size_t)b * 3 + d) * Mp + j] = xs * v;
  }
  const float* xb = xt1 + (size_t)b * 128 * N;
  float* fb = fbuf + (size_t)b * 256 * Mp;
  for (int c = 0; c < 128; ++c) fb[(size_t)c * Mp + j] = xb[(size_t)c * N + id] * v;
}

__global__ void aggregate_kernel(const float* xt1, const int* idxA, float* fbuf,
                                 int N, int Mp, int k) {
  int t = blockIdx.x * blockDim.x + threadIdx.x;
  int b = blockIdx.y;
  if (t >= 128 * Mp) return;
  int c = t / Mp, m = t - c * Mp;
  const float* xb = xt1 + (size_t)b * 128 * N + (size_t)c * N;
  const int* ib = idxA + ((size_t)b * Mp + m) * k;
  float mx = -3.4e38f;
  for (int j = 0; j < k; ++j) { float v = xb[ib[j]]; mx = v > mx ? v : mx; }
  fbuf[(size_t)b * 256 * Mp + (size_t)(128 + c) * Mp + m] = mx;
}

// ---------------------------------------------------------------------------
// 3-NN unpool; weights = softmax over RAW distances (reference quirk)
// ---------------------------------------------------------------------------
__global__ void unpool_kernel(const float* ns, const float* xyz, const float* feat,
                              float* out, int N, int Mp) {
  int n = blockIdx.x * blockDim.x + threadIdx.x;
  int b = blockIdx.y;
  if (n >= N) return;
  float q0 = xyz[(size_t)b * 3 * N + n];
  float q1 = xyz[(size_t)b * 3 * N + N + n];
  float q2 = xyz[(size_t)b * 3 * N + 2 * N + n];
  const float* nb = ns + (size_t)b * 3 * Mp;
  float bd[3] = {3.4e38f, 3.4e38f, 3.4e38f}; int bi[3] = {0, 0, 0};
  for (int m = 0; m < Mp; ++m) {
    float d0 = q0 - nb[m], d1 = q1 - nb[Mp + m], d2 = q2 - nb[2 * Mp + m];
    float d = d0 * d0 + d1 * d1 + d2 * d2;
    if (d < bd[2]) {
      if (d < bd[0])      { bd[2]=bd[1]; bi[2]=bi[1]; bd[1]=bd[0]; bi[1]=bi[0]; bd[0]=d; bi[0]=m; }
      else if (d < bd[1]) { bd[2]=bd[1]; bi[2]=bi[1]; bd[1]=d; bi[1]=m; }
      else                { bd[2]=d; bi[2]=m; }
    }
  }
  float mx = fmaxf(bd[0], fmaxf(bd[1], bd[2]));
  float e0 = __expf(bd[0] - mx), e1 = __expf(bd[1] - mx), e2 = __expf(bd[2] - mx);
  float inv = 1.f / (e0 + e1 + e2);
  float w0 = e0 * inv, w1 = e1 * inv, w2 = e2 * inv;
  const float* fb = feat + (size_t)b * 128 * Mp;
  float* ob = out + (size_t)b * 128 * N;
  for (int c = 0; c < 128; ++c)
    ob[(size_t)c * N + n] = w0 * fb[(size_t)c * Mp + bi[0]] +
                            w1 * fb[(size_t)c * Mp + bi[1]] +
                            w2 * fb[(size_t)c * Mp + bi[2]];
}

// ---------------------------------------------------------------------------
// small dense layers (classification head) and segmentation logits
// ---------------------------------------------------------------------------
__global__ void linear_kernel(const float* in, int Cin, const float* w, int Cout,
                              const float* preb, const float* g, const float* postb,
                              int lrelu, float* out, int B) {
  int t = blockIdx.x * blockDim.x + threadIdx.x;
  if (t >= B * Cout) return;
  int b = t / Cout, o = t - b * Cout;
  float s = 0.f;
  for (int c = 0; c < Cin; ++c) s += in[(size_t)b * Cin + c] * w[(size_t)o * Cin + c];
  if (preb)  s += preb[o];
  if (g)     s *= g[o] * BN_RSQRT;
  if (postb) s += postb[o];
  if (lrelu) s = (s >= 0.f) ? s : LRELU_SLOPE * s;
  out[(size_t)b * Cout + o] = s;
}

__global__ void seghead_kernel(const float* w10, const float* s, float* out, int N) {
  int t = blockIdx.x * blockDim.x + threadIdx.x;
  int b = blockIdx.y;
  if (t >= 2 * N) return;
  int o = t / N, n = t - o * N;
  const float* sb = s + (size_t)b * 128 * N;
  float acc = 0.f;
  for (int c = 0; c < 128; ++c) acc += w10[o * 128 + c] * sb[(size_t)c * N + n];
  out[((size_t)b * 2 + o) * N + n] = acc;
}

// ---------------------------------------------------------------------------
// Orchestration
// ---------------------------------------------------------------------------
extern "C" void kernel_launch(void* const* d_in, const int* in_sizes, int n_in,
                              void* d_out, int out_size, void* d_ws, size_t ws_size,
                              hipStream_t stream) {
  (void)in_sizes; (void)n_in; (void)out_size; (void)ws_size;
  const int B = 8, N = 2048, Mp = 512, KNN = 20;

  const float* x   = (const float*)d_in[0];
  const float* w1  = (const float*)d_in[1];  const float* g1  = (const float*)d_in[2];  const float* b1  = (const float*)d_in[3];
  const float* w2  = (const float*)d_in[4];  const float* g2  = (const float*)d_in[5];  const float* b2  = (const float*)d_in[6];
  const float* w2m = (const float*)d_in[7];  const float* g2m = (const float*)d_in[8];  const float* b2m = (const float*)d_in[9];
  const float* wp  = (const float*)d_in[10]; const float* bp  = (const float*)d_in[11];
  const float* w3  = (const float*)d_in[12]; const float* g3  = (const float*)d_in[13]; const float* b3  = (const float*)d_in[14];
  const float* w4  = (const float*)d_in[15]; const float* g4  = (const float*)d_in[16]; const float* b4  = (const float*)d_in[17];
  const float* w5  = (const float*)d_in[18]; const float* g5  = (const float*)d_in[19]; const float* b5  = (const float*)d_in[20];
  const float* wl1 = (const float*)d_in[21]; const float* gl1 = (const float*)d_in[22]; const float* bl1 = (const float*)d_in[23];
  const float* wl2 = (const float*)d_in[24]; const float* bl2v= (const float*)d_in[25]; const float* gl2 = (const float*)d_in[26]; const float* bl2 = (const float*)d_in[27];
  const float* wl3 = (const float*)d_in[28]; const float* bl3 = (const float*)d_in[29];
  const float* w6  = (const float*)d_in[30]; const float* g6  = (const float*)d_in[31]; const float* b6  = (const float*)d_in[32];
  const float* w7  = (const float*)d_in[33]; const float* g7  = (const float*)d_in[34]; const float* b7  = (const float*)d_in[35];
  const float* w8  = (const float*)d_in[36]; const float* g8  = (const float*)d_in[37]; const float* b8  = (const float*)d_in[38];
  const float* w9  = (const float*)d_in[39]; const float* g9  = (const float*)d_in[40]; const float* b9  = (const float*)d_in[41];
  const float* w10 = (const float*)d_in[42];

  // ---- workspace carve-up -------------------------------------------------
  char* wsp = (char*)d_ws;
  size_t off = 0;
  auto alloc = [&](size_t bytes) { void* p = wsp + off; off += (bytes + 255) & ~(size_t)255; return p; };
  int*      idx1   = (int*)     alloc((size_t)B * N * KNN * 4);
  float*    xt1    = (float*)   alloc((size_t)B * 128 * N * 4);
  unsigned* ubuf   = (unsigned*)alloc((size_t)B * 256 * Mp * 4);  // == B*64*N
  unsigned* uvec   = (unsigned*)alloc((size_t)B * 1024 * 4);
  float*    vecbuf = (float*)   alloc((size_t)B * 2048 * 4);
  float*    vecp   = (float*)   alloc((size_t)B * 1024 * 4);
  float*    fbuf   = (float*)   alloc((size_t)B * 256 * Mp * 4);
  float*    x34    = (float*)   alloc((size_t)B * 512 * Mp * 4);
  int*      idx3   = (int*)     alloc((size_t)B * Mp * 10 * 4);
  int*      idxA   = (int*)     alloc((size_t)B * Mp * KNN * 4);
  int*      pidx   = (int*)     alloc((size_t)B * Mp * 4);
  float*    pval   = (float*)   alloc((size_t)B * Mp * 4);
  float*    scbuf  = (float*)   alloc((size_t)B * N * 8 * 4);
  float*    sbufA  = (float*)   alloc((size_t)B * 128 * Mp * 4);
  float*    sbufB  = (float*)   alloc((size_t)B * 128 * Mp * 4);
  float*    sbufC  = (float*)   alloc((size_t)B * 128 * N * 4);
  float*    sbufD  = (float*)   alloc((size_t)B * 128 * N * 4);
  float*    h1     = (float*)   alloc((size_t)B * 512 * 4);
  float*    h2     = (float*)   alloc((size_t)B * 256 * 4);
  unsigned short* Wbf = (unsigned short*)alloc((size_t)1500000 * 2);

  float* out      = (float*)d_out;
  float* out_cls  = out;                       // (8,15)
  float* out_seg  = out + 120;                 // (8,2,2048)
  float* out_node = out + 120 + 2 * N * B;     // (8,3,512)
  float* out_ns   = out_node + 3 * Mp * B;     // (8,3,512)

  dim3 blk256(256), blk128(128);
  GemmP p;

  // ---- pre-convert conv weights to bf16 (16B-aligned regions) ------------
  size_t wboff = 0;
  auto cvt = [&](const float* src, int n) {
    unsigned short* d = Wbf + wboff;
    wboff += ((size_t)n + 7) & ~(size_t)7;
    cvt_bf16_kernel<<<(n + 255) / 256, blk256, 0, stream>>>(src, d, n);
    return (const unsigned short*)d;
  };
  const unsigned short* w1b  = cvt(w1, 64 * 6);
  const unsigned short* w2b  = cvt(w2, 64 * 128);
  const unsigned short* w2mb = cvt(w2m, 1024 * 128);
  const unsigned short* wpb  = cvt(wp, 1024 * 128);
  const unsigned short* w3b  = cvt(w3, 256 * 512);
  const unsigned short* w4b  = cvt(w4, 256 * 512);
  const unsigned short* w5b  = cvt(w5, 1024 * 512);
  const unsigned short* w6b  = cvt(w6, 128 * 2304);
  const unsigned short* w7b  = cvt(w7, 128 * 384);
  const unsigned short* w8b  = cvt(w8, 128 * 192);
  const unsigned short* w9b  = cvt(w9, 128 * 192);

  // ================= EdgeConv 1 (x -> x1) =================================
  knn_kernel<<<dim3((N + 127) / 128, B), blk128, 0, stream>>>(x, 3 * N, x, 3 * N, 3, N, N, KNN, idx1);
  hipMemsetAsync(ubuf, 0, (size_t)B * 64 * N * 4, stream);
  p = {}; p.W = w1b; p.gamma = g1; p.beta = b1; p.Cout = 64; p.Cin = 6; p.Ncols = N * KNN;
  p.X0 = x; p.x0Bs = 3 * N; p.idx = idx1; p.kk = KNN; p.Nq = N; p.Cedge = 3; p.Npts = N;
  p.uout = ubuf; p.outN = N;
  gemm_wmma<1, 1, true><<<dim3(p.Ncols / 64, 1, B), blk256, 0, stream>>>(p);
  decode_kernel<<<dim3((64 * N + 255) / 256, B), blk256, 0, stream>>>(ubuf, xt1, 64, N, 128, 0);

  // ================= EdgeConv 2 (x1 -> x2) ================================
  knn_kernel<<<dim3((N + 127) / 128, B), blk128, 0, stream>>>(xt1, 128 * N, xt1, 128 * N, 64, N, N, KNN, idx1);
  hipMemsetAsync(ubuf, 0, (size_t)B * 64 * N * 4, stream);
  p = {}; p.W = w2b; p.gamma = g2; p.beta = b2; p.Cout = 64; p.Cin = 128; p.Ncols = N * KNN;
  p.X0 = xt1; p.x0Bs = 128 * N; p.idx = idx1; p.kk = KNN; p.Nq = N; p.Cedge = 64; p.Npts = N;
  p.uout = ubuf; p.outN = N;
  gemm_wmma<1, 1, true><<<dim3(p.Ncols / 64, 1, B), blk256, 0, stream>>>(p);
  decode_kernel<<<dim3((64 * N + 255) / 256, B), blk256, 0, stream>>>(ubuf, xt1, 64, N, 128, 64);

  // ================= x_t1 embedding, fused max over N -> vector[:1024] ====
  hipMemsetAsync(uvec, 0, (size_t)B * 1024 * 4, stream);
  p = {}; p.W = w2mb; p.gamma = g2m; p.beta = b2m; p.Cout = 1024; p.Cin = 128; p.Ncols = N;
  p.X0 = xt1; p.x0Bs = 128 * N; p.uout = uvec;
  gemm_wmma<0, 2, true><<<dim3(N / 64, 16, B), blk256, 0, stream>>>(p);
  decode_kernel<<<dim3((1024 + 255) / 256, B), blk256, 0, stream>>>(uvec, vecbuf, 1024, 1, 2048, 0);

  // ================= Pool: proj max (no BN/lrelu) -> vecp =================
  hipMemsetAsync(uvec, 0, (size_t)B * 1024 * 4, stream);
  p = {}; p.W = wpb; p.gamma = nullptr; p.beta = bp; p.Cout = 1024; p.Cin = 128; p.Ncols = N;
  p.X0 = xt1; p.x0Bs = 128 * N; p.uout = uvec;
  gemm_wmma<0, 2, false><<<dim3(N / 64, 16, B), blk256, 0, stream>>>(p);
  decode_kernel<<<dim3((1024 + 255) / 256, B), blk256, 0, stream>>>(uvec, vecp, 1024, 1, 1024, 0);

  scores_kernel<<<dim3((N * 8 + 255) / 256, B), blk256, 0, stream>>>(xt1, vecp, scbuf, N);
  pool_topk_kernel<<<B, 8, 0, stream>>>(scbuf, pidx, pval, N, Mp / 8);
  pool_gather_kernel<<<dim3((Mp + 127) / 128, B), blk128, 0, stream>>>(x, xt1, pidx, pval,
                                                                       out_node, out_ns, fbuf, N, Mp);
  // aggregate: kNN(node -> xyz), max-gather of x_t1_ features
  knn_kernel<<<dim3((Mp + 127) / 128, B), blk128, 0, stream>>>(out_node, 3 * Mp, x, 3 * N, 3, Mp, N, KNN, idxA);
  aggregate_kernel<<<dim3((128 * Mp + 255) / 256, B), blk256, 0, stream>>>(xt1, idxA, fbuf, N, Mp, KNN);

  // ================= EdgeConv 3 (f -> x3) =================================
  knn_kernel<<<dim3((Mp + 127) / 128, B), blk128, 0, stream>>>(fbuf, 256 * Mp, fbuf, 256 * Mp, 256, Mp, Mp, 10, idx3);
  hipMemsetAsync(ubuf, 0, (size_t)B * 256 * Mp * 4, stream);
  p = {}; p.W = w3b; p.gamma = g3; p.beta = b3; p.Cout = 256; p.Cin = 512; p.Ncols = Mp * 10;
  p.X0 = fbuf; p.x0Bs = 256 * Mp; p.idx = idx3; p.kk = 10; p.Nq = Mp; p.Cedge = 256; p.Npts = Mp;
  p.uout = ubuf; p.outN = Mp;
  gemm_wmma<1, 1, true><<<dim3(p.Ncols / 64, 4, B), blk256, 0, stream>>>(p);
  decode_kernel<<<dim3((256 * Mp + 255) / 256, B), blk256, 0, stream>>>(ubuf, x34, 256, Mp, 512, 0);

  // ================= EdgeConv 4 (x3 -> x4) ================================
  knn_kernel<<<dim3((Mp + 127) / 128, B), blk128, 0, stream>>>(x34, 512 * Mp, x34, 512 * Mp, 256, Mp, Mp, 10, idx3);
  hipMemsetAsync(ubuf, 0, (size_t)B * 256 * Mp * 4, stream);
  p = {}; p.W = w4b; p.gamma = g4; p.beta = b4; p.Cout = 256; p.Cin = 512; p.Ncols = Mp * 10;
  p.X0 = x34; p.x0Bs = 512 * Mp; p.idx = idx3; p.kk = 10; p.Nq = Mp; p.Cedge = 256; p.Npts = Mp;
  p.uout = ubuf; p.outN = Mp;
  gemm_wmma<1, 1, true><<<dim3(p.Ncols / 64, 4, B), blk256, 0, stream>>>(p);
  decode_kernel<<<dim3((256 * Mp + 255) / 256, B), blk256, 0, stream>>>(ubuf, x34, 256, Mp, 512, 256);

  // ================= x_t2 embedding, fused max over M -> vector[1024:] ====
  hipMemsetAsync(uvec, 0, (size_t)B * 1024 * 4, stream);
  p = {}; p.W = w5b; p.gamma = g5; p.beta = b5; p.Cout = 1024; p.Cin = 512; p.Ncols = Mp;
  p.X0 = x34; p.x0Bs = 512 * Mp; p.uout = uvec;
  gemm_wmma<0, 2, true><<<dim3(Mp / 64, 16, B), blk256, 0, stream>>>(p);
  decode_kernel<<<dim3((1024 + 255) / 256, B), blk256, 0, stream>>>(uvec, vecbuf, 1024, 1, 2048, 1024);

  // ================= classification head ==================================
  linear_kernel<<<(B * 512 + 255) / 256, blk256, 0, stream>>>(vecbuf, 2048, wl1, 512, nullptr, gl1, bl1, 1, h1, B);
  linear_kernel<<<(B * 256 + 255) / 256, blk256, 0, stream>>>(h1, 512, wl2, 256, bl2v, gl2, bl2, 1, h2, B);
  linear_kernel<<<(B * 15 + 255) / 256, blk256, 0, stream>>>(h2, 256, wl3, 15, bl3, nullptr, nullptr, 0, out_cls, B);

  // ================= segmentation decoder =================================
  // conv6: concat(broadcast vector(2048), x4(256)) -> 128
  p = {}; p.W = w6b; p.gamma = g6; p.beta = b6; p.Cout = 128; p.Cin = 2304; p.Ncols = Mp;
  p.X0 = vecbuf; p.x0Bs = 2048; p.bc1 = 1; p.C1 = 2048;
  p.X1 = x34 + (size_t)256 * Mp; p.x1Bs = 512 * Mp; p.C2 = 256;
  p.out = sbufA; p.outCB = 128; p.chOff = 0;
  gemm_wmma<2, 0, true><<<dim3(Mp / 64, 2, B), blk256, 0, stream>>>(p);
  // conv7: concat(s(128), x3(256)) -> 128
  p = {}; p.W = w7b; p.gamma = g7; p.beta = b7; p.Cout = 128; p.Cin = 384; p.Ncols = Mp;
  p.X0 = sbufA; p.x0Bs = 128 * Mp; p.bc1 = 0; p.C1 = 128;
  p.X1 = x34; p.x1Bs = 512 * Mp; p.C2 = 256;
  p.out = sbufB; p.outCB = 128; p.chOff = 0;
  gemm_wmma<2, 0, true><<<dim3(Mp / 64, 2, B), blk256, 0, stream>>>(p);
  // unpool to full resolution
  unpool_kernel<<<dim3((N + 127) / 128, B), blk128, 0, stream>>>(out_ns, x, sbufB, sbufC, N, Mp);
  // conv8: concat(s_up(128), x2(64)) -> 128
  p = {}; p.W = w8b; p.gamma = g8; p.beta = b8; p.Cout = 128; p.Cin = 192; p.Ncols = N;
  p.X0 = sbufC; p.x0Bs = 128 * N; p.bc1 = 0; p.C1 = 128;
  p.X1 = xt1 + (size_t)64 * N; p.x1Bs = 128 * N; p.C2 = 64;
  p.out = sbufD; p.outCB = 128; p.chOff = 0;
  gemm_wmma<2, 0, true><<<dim3(N / 64, 2, B), blk256, 0, stream>>>(p);
  // conv9: concat(s(128), x1(64)) -> 128
  p = {}; p.W = w9b; p.gamma = g9; p.beta = b9; p.Cout = 128; p.Cin = 192; p.Ncols = N;
  p.X0 = sbufD; p.x0Bs = 128 * N; p.bc1 = 0; p.C1 = 128;
  p.X1 = xt1; p.x1Bs = 128 * N; p.C2 = 64;
  p.out = sbufC; p.outCB = 128; p.chOff = 0;
  gemm_wmma<2, 0, true><<<dim3(N / 64, 2, B), blk256, 0, stream>>>(p);
  // final 2-class logits
  seghead_kernel<<<dim3((2 * N + 255) / 256, B), blk256, 0, stream>>>(w10, sbufC, out_seg, N);
}
